// GPT2FetchingConditioner_32822140076236
// MI455X (gfx1250) — matrix operand order repaired and checked by reading, not compiled, not run on hardware
//
#include <hip/hip_runtime.h>
#include <hip/hip_bf16.h>
#include <cstdint>
#include <cstddef>

// ---------------------------------------------------------------------------
// CDNA5 (gfx1250) wave32 WMMA types
// ---------------------------------------------------------------------------
typedef __attribute__((ext_vector_type(16))) __bf16 v16bf;
typedef __attribute__((ext_vector_type(8)))  float  v8f;

union FragU {
    v16bf v;
    unsigned int u[8];
    unsigned short us[16];
};

__device__ __forceinline__ unsigned short f2b(float f) {
    union { float f; unsigned int u; } c; c.f = f;
    unsigned int r = c.u + 0x7FFFu + ((c.u >> 16) & 1u);   // round-to-nearest-even
    return (unsigned short)(r >> 16);
}

// interleave two bf16x2 dwords (rows k and k+1, same two columns) into
// column-major k-pairs: out0 = (a.lo, b.lo), out1 = (a.hi, b.hi)
__device__ __forceinline__ void kpair(unsigned a, unsigned b,
                                      unsigned &o0, unsigned &o1) {
    o0 = (a & 0xFFFFu) | (b << 16);
    o1 = (a >> 16) | (b & 0xFFFF0000u);
}

// CDNA5 async memory->LDS copy (ASYNCcnt tracked), 16B per lane
#define USE_ASYNC_KTILE 1
__device__ __forceinline__ void async_load_b128(unsigned lds_off, const void* gaddr) {
    asm volatile("global_load_async_to_lds_b128 %0, %1, off"
                 :: "v"(lds_off), "v"(gaddr) : "memory");
}
__device__ __forceinline__ void wait_asynccnt0() {
    asm volatile("s_wait_asynccnt 0x0" ::: "memory");
}

// ---------------------------------------------------------------------------
// f32 -> bf16 conversion (grid-stride)
// ---------------------------------------------------------------------------
__global__ void cvt_f32_bf16(const float* __restrict__ in,
                             unsigned short* __restrict__ out, size_t n) {
    size_t i = (size_t)blockIdx.x * blockDim.x + threadIdx.x;
    size_t stride = (size_t)gridDim.x * blockDim.x;
    for (; i < n; i += stride) out[i] = f2b(in[i]);
}

// ---------------------------------------------------------------------------
// Tiled bf16 GEMM with WMMA: C[M,N] = A[M,K] * B[K,N] (+bias)(+res)(+gelu)
// Block tile 128x128, BK=32. 256 threads = 8 waves; wave tile 32x64 = 2x4
// accumulators of v_wmma_f32_16x16x32_bf16.
// B tile is stored in LDS as k-pair-packed dwords so B fragments are plain
// dword loads (no scalar u16 traffic in the hot loop).
// ---------------------------------------------------------------------------
template<bool BIAS, bool RES, bool GELU_, bool OUTB>
__global__ void __launch_bounds__(256) gemm_bf16_wmma(
    const unsigned short* __restrict__ A,
    const unsigned short* __restrict__ Bw,
    const float* __restrict__ bias,
    const float* __restrict__ res,
    float* __restrict__ Cf,
    unsigned short* __restrict__ Cb,
    int Mdim, int Ndim, int Kdim)
{
    __shared__ unsigned short As[128][40];    // 128 rows x 32 k (pad->40)
    __shared__ unsigned int   BsP[16][132];   // k-pair kp=0..15 x 128 cols (pad->132)

    const int tid  = threadIdx.x;
    const int wave = tid >> 5;
    const int lane = tid & 31;
    const int half = lane >> 4;
    const int l16  = lane & 15;
    const int wm   = wave & 3;     // 4 waves along M
    const int wn   = wave >> 2;    // 2 waves along N
    const int m0   = blockIdx.y * 128;
    const int n0   = blockIdx.x * 128;

    // B-tile loader mapping: one (k-pair, 8-col chunk) unit per thread
    const int kp  = tid >> 4;          // 0..15
    const int c8  = (tid & 15) * 8;    // 0..120

    v8f zero = {};
    v8f acc[2][4];
#pragma unroll
    for (int i = 0; i < 2; ++i)
#pragma unroll
        for (int j = 0; j < 4; ++j) acc[i][j] = zero;

    for (int k0 = 0; k0 < Kdim; k0 += 32) {
        // ---- A tile: 2 coalesced 16B chunks per thread
#pragma unroll
        for (int c = 0; c < 2; ++c) {
            int lin = (tid * 2 + c) * 8;
            int ra = lin >> 5, ca = lin & 31;
            *(uint4*)&As[ra][ca] =
                *(const uint4*)&A[(size_t)(m0 + ra) * Kdim + k0 + ca];
        }
        // ---- B tile: rows 2kp and 2kp+1, interleave into k-pair dwords
        {
            const unsigned short* gB =
                &Bw[(size_t)(k0 + 2 * kp) * Ndim + n0 + c8];
            uint4 rA = *(const uint4*)gB;
            uint4 rB = *(const uint4*)(gB + Ndim);
            uint4 o0, o1;
            kpair(rA.x, rB.x, o0.x, o0.y);
            kpair(rA.y, rB.y, o0.z, o0.w);
            kpair(rA.z, rB.z, o1.x, o1.y);
            kpair(rA.w, rB.w, o1.z, o1.w);
            *(uint4*)&BsP[kp][c8]     = o0;
            *(uint4*)&BsP[kp][c8 + 4] = o1;
        }
        // prefetch next k-tile into GL2 while this one is consumed
        if (k0 + 32 < Kdim) {
            __builtin_prefetch(&A[(size_t)(m0 + (tid >> 1)) * Kdim + k0 + 32], 0, 1);
            __builtin_prefetch(&Bw[(size_t)(k0 + 32 + kp) * Ndim + n0 + c8], 0, 1);
        }
        __syncthreads();

        // A fragments: 16x32 bf16, lane l16 = row, K split across half-waves
        FragU af[2];
#pragma unroll
        for (int i = 0; i < 2; ++i) {
            int row = wm * 32 + i * 16 + l16;
#pragma unroll
            for (int v = 0; v < 8; ++v) {
                int kk = half * 16 + (v << 1);
                af[i].u[v] = *(const unsigned int*)&As[row][kk];
            }
        }
        // B fragments: dword per k-pair, lane l16 = column
        FragU bfr[4];
#pragma unroll
        for (int j = 0; j < 4; ++j) {
            int col = wn * 64 + j * 16 + l16;
#pragma unroll
            for (int v = 0; v < 8; ++v)
                bfr[j].u[v] = BsP[half * 8 + v][col];
        }
#pragma unroll
        for (int i = 0; i < 2; ++i)
#pragma unroll
            for (int j = 0; j < 4; ++j)
                acc[i][j] = __builtin_amdgcn_wmma_f32_16x16x32_bf16(
                    false, af[i].v, false, bfr[j].v, (short)0, acc[i][j],
                    false, false);
        __syncthreads();
    }

    // epilogue: C layout = lane l16 column, VGPR r = row r (+8 for upper half)
#pragma unroll
    for (int j = 0; j < 4; ++j) {
        int col = n0 + wn * 64 + j * 16 + l16;
        float bv = BIAS ? bias[col] : 0.0f;
#pragma unroll
        for (int i = 0; i < 2; ++i) {
#pragma unroll
            for (int r = 0; r < 8; ++r) {
                int row = m0 + wm * 32 + i * 16 + r + half * 8;
                size_t idx = (size_t)row * Ndim + col;
                float x = acc[i][j][r] + bv;
                if (RES)   x += res[idx];
                if (GELU_) x = 0.5f * x * (1.0f + erff(x * 0.70710678118f));
                if (OUTB)  Cb[idx] = f2b(x);
                else       Cf[idx] = x;
            }
        }
    }
}

// ---------------------------------------------------------------------------
// Causal flash attention with WMMA.  One block per (q-tile of 64 rows, h, b).
// 128 threads = 4 waves, each wave owns 16 q rows.  DH = 64.
// K tile fetched with CDNA5 async memory->LDS copies; V tile stored k-pair
// packed so the P*V B-fragments are plain dword loads.
// ---------------------------------------------------------------------------
__global__ void __launch_bounds__(128) flash_attn_wmma(
    const unsigned short* __restrict__ Q,
    const unsigned short* __restrict__ Kt,
    const unsigned short* __restrict__ Vt,
    unsigned short* __restrict__ O,
    int S, int Dm)
{
    __shared__ unsigned short Ks[64][72];     // row-major keys (pad 64->72)
    __shared__ unsigned int   VsP[32][68];    // k-pair packed V (pad 64->68)
    __shared__ unsigned short Ps[4][16][72];  // per-wave P staging (C->A relayout)

    const int qi = blockIdx.x, h = blockIdx.y, b = blockIdx.z;
    const int tid  = threadIdx.x;
    const int wave = tid >> 5, lane = tid & 31;
    const int half = lane >> 4, l16 = lane & 15;

    const size_t base = (size_t)b * S * Dm + (size_t)h * 64;

    // Q fragments (A layout), loaded once: 2 k-steps over DH=64
    FragU qf[2];
    {
        int qrow = qi * 64 + wave * 16 + l16;
#pragma unroll
        for (int kk = 0; kk < 2; ++kk)
#pragma unroll
            for (int v = 0; v < 8; ++v) {
                int dh = kk * 32 + half * 16 + (v << 1);
                qf[kk].u[v] = *(const unsigned int*)&Q[base + (size_t)qrow * Dm + dh];
            }
    }

    v8f zero = {};
    v8f o[4];
#pragma unroll
    for (int j2 = 0; j2 < 4; ++j2) o[j2] = zero;
    float mrow[8], lrow[8];
#pragma unroll
    for (int r = 0; r < 8; ++r) { mrow[r] = -3.0e38f; lrow[r] = 0.0f; }

    const int ntile = qi + 1;   // causal: key tiles 0..qi
    for (int jt = 0; jt < ntile; ++jt) {
        __syncthreads();        // protect LDS reuse across waves
        const int kbase = jt * 64;

        // ---- K tile via async memory->LDS (ASYNCcnt), 16B per lane x4
#if USE_ASYNC_KTILE
#pragma unroll
        for (int c = 0; c < 4; ++c) {
            int lin = (tid * 4 + c) * 8;
            int r = lin >> 6, cc = lin & 63;
            unsigned loff = (unsigned)(size_t)&Ks[r][cc];
            async_load_b128(loff, &Kt[base + (size_t)(kbase + r) * Dm + cc]);
        }
#else
#pragma unroll
        for (int c = 0; c < 4; ++c) {
            int lin = (tid * 4 + c) * 8;
            int r = lin >> 6, cc = lin & 63;
            *(uint4*)&Ks[r][cc] =
                *(const uint4*)&Kt[base + (size_t)(kbase + r) * Dm + cc];
        }
#endif
        // ---- V tile: interleave rows 2kp,2kp+1 into k-pair dwords
#pragma unroll
        for (int c = 0; c < 2; ++c) {
            int u = tid * 2 + c;              // 0..255 units
            int kp = u >> 3;                  // 0..31
            int dc = (u & 7) * 8;             // 0..56
            const unsigned short* gV =
                &Vt[base + (size_t)(kbase + 2 * kp) * Dm + dc];
            uint4 rA = *(const uint4*)gV;
            uint4 rB = *(const uint4*)(gV + Dm);
            uint4 o0, o1;
            kpair(rA.x, rB.x, o0.x, o0.y);
            kpair(rA.y, rB.y, o0.z, o0.w);
            kpair(rA.z, rB.z, o1.x, o1.y);
            kpair(rA.w, rB.w, o1.z, o1.w);
            *(uint4*)&VsP[kp][dc]     = o0;
            *(uint4*)&VsP[kp][dc + 4] = o1;
        }
#if USE_ASYNC_KTILE
        wait_asynccnt0();
#endif
        __syncthreads();

        // S = Q * K^T  (B frag: column = key, k = dh; K row-major dh-contiguous)
        v8f s[4];
#pragma unroll
        for (int j2 = 0; j2 < 4; ++j2) {
            s[j2] = zero;
#pragma unroll
            for (int kk = 0; kk < 2; ++kk) {
                FragU bf;
                int key = j2 * 16 + l16;
#pragma unroll
                for (int v = 0; v < 8; ++v) {
                    int dh = kk * 32 + half * 16 + v * 2;
                    bf.u[v] = *(const unsigned int*)&Ks[key][dh];
                }
                s[j2] = __builtin_amdgcn_wmma_f32_16x16x32_bf16(
                    false, qf[kk].v, false, bf.v, (short)0, s[j2], false, false);
            }
        }

        // scale + causal mask + online softmax statistics
        float mnew[8];
#pragma unroll
        for (int r = 0; r < 8; ++r) mnew[r] = -3.0e38f;
#pragma unroll
        for (int j2 = 0; j2 < 4; ++j2)
#pragma unroll
            for (int r = 0; r < 8; ++r) {
                float x = s[j2][r] * 0.125f;             // 1/sqrt(64)
                int qr = qi * 64 + wave * 16 + r + half * 8;
                int kc = kbase + j2 * 16 + l16;
                if (kc > qr) x = -3.0e38f;
                s[j2][r] = x;
                mnew[r] = fmaxf(mnew[r], x);
            }
        float alpha[8];
#pragma unroll
        for (int r = 0; r < 8; ++r) {
#pragma unroll
            for (int d = 1; d < 16; d <<= 1)
                mnew[r] = fmaxf(mnew[r], __shfl_xor(mnew[r], d, 32));
            float mf = fmaxf(mrow[r], mnew[r]);
            alpha[r] = __expf(mrow[r] - mf);
            mrow[r] = mf;
        }
        float rsum[8] = {};
#pragma unroll
        for (int j2 = 0; j2 < 4; ++j2)
#pragma unroll
            for (int r = 0; r < 8; ++r) {
                float p = __expf(s[j2][r] - mrow[r]);
                s[j2][r] = p;
                rsum[r] += p;
            }
#pragma unroll
        for (int r = 0; r < 8; ++r) {
#pragma unroll
            for (int d = 1; d < 16; d <<= 1)
                rsum[r] += __shfl_xor(rsum[r], d, 32);
            lrow[r] = lrow[r] * alpha[r] + rsum[r];
        }
#pragma unroll
        for (int j2 = 0; j2 < 4; ++j2)
#pragma unroll
            for (int r = 0; r < 8; ++r) o[j2][r] *= alpha[r];

        // P: C layout -> bf16 A layout via per-wave LDS region
#pragma unroll
        for (int j2 = 0; j2 < 4; ++j2)
#pragma unroll
            for (int r = 0; r < 8; ++r)
                Ps[wave][r + half * 8][j2 * 16 + l16] = f2b(s[j2][r]);

        // O += P * V  (A frag from Ps; B frag: dword per key-pair, col = dh)
#pragma unroll
        for (int kk = 0; kk < 2; ++kk) {
            FragU pa;
#pragma unroll
            for (int v = 0; v < 8; ++v) {
                int kkey = kk * 32 + half * 16 + (v << 1);
                pa.u[v] = *(const unsigned int*)&Ps[wave][l16][kkey];
            }
#pragma unroll
            for (int j2 = 0; j2 < 4; ++j2) {
                FragU vf;
                int dh = j2 * 16 + l16;
#pragma unroll
                for (int v = 0; v < 8; ++v)
                    vf.u[v] = VsP[kk * 16 + half * 8 + v][dh];
                o[j2] = __builtin_amdgcn_wmma_f32_16x16x32_bf16(
                    false, pa.v, false, vf.v, (short)0, o[j2], false, false);
            }
        }
    }

    // normalize and store bf16 output
#pragma unroll
    for (int j2 = 0; j2 < 4; ++j2)
#pragma unroll
        for (int r = 0; r < 8; ++r) {
            int qr = qi * 64 + wave * 16 + r + half * 8;
            int dh = j2 * 16 + l16;
            float val = o[j2][r] / lrow[r];
            O[base + (size_t)qr * Dm + dh] = f2b(val);
        }
}

// ---------------------------------------------------------------------------
// LayerNorm over last dim D; writes f32 and bf16 copies.
// ---------------------------------------------------------------------------
__global__ void __launch_bounds__(256) layernorm_k(
    const float* __restrict__ in, const float* __restrict__ g,
    const float* __restrict__ bta, float* __restrict__ out,
    unsigned short* __restrict__ outb, int D)
{
    __shared__ float red[256], red2[256];
    const int row = blockIdx.x, tid = threadIdx.x;
    const float* p = in + (size_t)row * D;
    float s = 0.0f, s2 = 0.0f;
    for (int i = tid; i < D; i += 256) { float v = p[i]; s += v; s2 += v * v; }
    red[tid] = s; red2[tid] = s2;
    __syncthreads();
    for (int st = 128; st > 0; st >>= 1) {
        if (tid < st) { red[tid] += red[tid + st]; red2[tid] += red2[tid + st]; }
        __syncthreads();
    }
    float mean = red[0] / (float)D;
    float var  = red2[0] / (float)D - mean * mean;
    float rstd = rsqrtf(var + 1e-5f);
    for (int i = tid; i < D; i += 256) {
        float v = (p[i] - mean) * rstd * g[i] + bta[i];
        size_t idx = (size_t)row * D + i;
        out[idx]  = v;
        outb[idx] = f2b(v);
    }
}

// ---------------------------------------------------------------------------
// Host orchestration
// ---------------------------------------------------------------------------
extern "C" void kernel_launch(void* const* d_in, const int* in_sizes, int n_in,
                              void* d_out, int out_size, void* d_ws, size_t ws_size,
                              hipStream_t stream)
{
    (void)in_sizes; (void)n_in; (void)out_size; (void)ws_size;
    constexpr int Bb = 4, S = 2048, D = 768, Hh = 12, F = 3072, L = 4;
    constexpr int M  = Bb * S;   // 8192 token rows

    const float* x_in = (const float*)d_in[0];
    // d_in[1] = attn_mask (pure causal -> implied in-kernel)
    const float* Wq  = (const float*)d_in[2];
    const float* Wk  = (const float*)d_in[3];
    const float* Wv  = (const float*)d_in[4];
    const float* Wo  = (const float*)d_in[5];
    const float* bo  = (const float*)d_in[6];
    const float* g1  = (const float*)d_in[7];
    const float* be1 = (const float*)d_in[8];
    const float* W1  = (const float*)d_in[9];
    const float* b1  = (const float*)d_in[10];
    const float* W2  = (const float*)d_in[11];
    const float* b2  = (const float*)d_in[12];
    const float* g2  = (const float*)d_in[13];
    const float* be2 = (const float*)d_in[14];

    char* wsp = (char*)d_ws;
    auto carve = [&](size_t bytes) -> void* {
        void* p = (void*)wsp;
        wsp += (bytes + 255) & ~(size_t)255;
        return p;
    };
    float* xf = (float*)carve((size_t)M * D * 4);
    float* yf = (float*)carve((size_t)M * D * 4);
    float* hf = (float*)carve((size_t)M * D * 4);
    unsigned short* xb  = (unsigned short*)carve((size_t)M * D * 2);
    unsigned short* hb  = (unsigned short*)carve((size_t)M * D * 2);
    unsigned short* qb  = (unsigned short*)carve((size_t)M * D * 2);
    unsigned short* kb  = (unsigned short*)carve((size_t)M * D * 2);
    unsigned short* vb  = (unsigned short*)carve((size_t)M * D * 2);
    unsigned short* avb = (unsigned short*)carve((size_t)M * D * 2);
    unsigned short* f1b = (unsigned short*)carve((size_t)M * F * 2);
    unsigned short* wqb = (unsigned short*)carve((size_t)D * D * 2);
    unsigned short* wkb = (unsigned short*)carve((size_t)D * D * 2);
    unsigned short* wvb = (unsigned short*)carve((size_t)D * D * 2);
    unsigned short* wob = (unsigned short*)carve((size_t)D * D * 2);
    unsigned short* w1b = (unsigned short*)carve((size_t)D * F * 2);
    unsigned short* w2b = (unsigned short*)carve((size_t)F * D * 2);

    auto cvt = [&](const float* src, unsigned short* dst, size_t n) {
        cvt_f32_bf16<<<1024, 256, 0, stream>>>(src, dst, n);
    };

    const dim3 blk256(256), blk128(128);
    const dim3 gD(D / 128, M / 128);     // GEMMs with N = D
    const dim3 gF(F / 128, M / 128);     // FFN1 with N = F
    const dim3 gAttn(S / 64, Hh, Bb);

    for (int l = 0; l < L; ++l) {
        const float* curx = (l == 0) ? x_in : xf;
        if (l == 0) cvt(x_in, xb, (size_t)M * D);

        cvt(Wq + (size_t)l * D * D, wqb, (size_t)D * D);
        cvt(Wk + (size_t)l * D * D, wkb, (size_t)D * D);
        cvt(Wv + (size_t)l * D * D, wvb, (size_t)D * D);
        cvt(Wo + (size_t)l * D * D, wob, (size_t)D * D);
        cvt(W1 + (size_t)l * D * F, w1b, (size_t)D * F);
        cvt(W2 + (size_t)l * F * D, w2b, (size_t)F * D);

        // QKV projections (bf16 out)
        gemm_bf16_wmma<false, false, false, true><<<gD, blk256, 0, stream>>>(
            xb, wqb, nullptr, nullptr, nullptr, qb, M, D, D);
        gemm_bf16_wmma<false, false, false, true><<<gD, blk256, 0, stream>>>(
            xb, wkb, nullptr, nullptr, nullptr, kb, M, D, D);
        gemm_bf16_wmma<false, false, false, true><<<gD, blk256, 0, stream>>>(
            xb, wvb, nullptr, nullptr, nullptr, vb, M, D, D);

        // causal multi-head attention
        flash_attn_wmma<<<gAttn, blk128, 0, stream>>>(qb, kb, vb, avb, S, D);

        // output projection + bias + residual (f32)
        gemm_bf16_wmma<true, true, false, false><<<gD, blk256, 0, stream>>>(
            avb, wob, bo + (size_t)l * D, curx, yf, nullptr, M, D, D);

        // LN1 -> h (f32) and hb (bf16)
        layernorm_k<<<M, blk256, 0, stream>>>(
            yf, g1 + (size_t)l * D, be1 + (size_t)l * D, hf, hb, D);

        // FFN1 + bias + exact GELU (bf16 out)
        gemm_bf16_wmma<true, false, true, true><<<gF, blk256, 0, stream>>>(
            hb, w1b, b1 + (size_t)l * F, nullptr, nullptr, f1b, M, F, D);

        // FFN2 + bias + residual h (f32)
        gemm_bf16_wmma<true, true, false, false><<<gD, blk256, 0, stream>>>(
            f1b, w2b, b2 + (size_t)l * D, hf, yf, nullptr, M, D, F);

        // LN2 -> next x (f32; d_out on last layer) and xb (bf16 for next QKV)
        float* outp = (l == L - 1) ? (float*)d_out : xf;
        layernorm_k<<<M, blk256, 0, stream>>>(
            yf, g2 + (size_t)l * D, be2 + (size_t)l * D, outp, xb, D);
    }
}
